// Qwen3MoeSparseMoeBlock_50268297232748
// MI455X (gfx1250) — matrix operand order repaired and checked by reading, not compiled, not run on hardware
//
#include <hip/hip_runtime.h>
#include <hip/hip_bf16.h>

typedef __attribute__((ext_vector_type(16))) __bf16 v16bf;
typedef __attribute__((ext_vector_type(8)))  __bf16 v8bf;
typedef __attribute__((ext_vector_type(8)))  float  v8f;

#define DH    2048   // hidden size
#define DI    768    // intermediate size
#define NEXP  32     // experts
#define NTOK  1024   // tokens
#define NTOPK 4
#define LDH   776    // padded bf16 row stride for h tile in LDS (bank-conflict free)

// ---------------------------------------------------------------- WMMA helpers

__device__ __forceinline__ v8f wmma_bf16(v16bf a, v16bf b, v8f c) {
  // D = A(16x32 bf16) * B(32x16 bf16) + C(16x16 f32)
  return __builtin_amdgcn_wmma_f32_16x16x32_bf16(false, a, false, b, (short)0, c,
                                                 false, false);
}

// Raw (pre-conversion) fp32 B fragment: 16 contiguous floats of a W row.
// B = W^T with W row-major [N][K]; lane l: col n = l&15 (= W row), g = l>>4;
// holds K = kb + g*16 + (0..15). rowPtrG16 includes (n*ld + g*16).
struct RawB { v8f c0, c1; };

__device__ __forceinline__ RawB loadB_raw(const float* __restrict__ rowPtrG16, int kb) {
  RawB r;
  r.c0 = *(const v8f*)(rowPtrG16 + kb);
  r.c1 = *(const v8f*)(rowPtrG16 + kb + 8);
  return r;
}

__device__ __forceinline__ v16bf cvtB(RawB r) {
  v16bf o;
#pragma unroll
  for (int i = 0; i < 8; ++i) { o[i] = (__bf16)r.c0[i]; o[i + 8] = (__bf16)r.c1[i]; }
  return o;
}

// A fragment (16x32 bf16) from bf16 row-major source (global xbf or LDS h tile).
// lane l: row = l&15, g = l>>4; K = {kb+g*16 .. +15}.
// rowPtrG8 must already include (row*ld + g*16).
__device__ __forceinline__ v16bf fragA_bf16(const __bf16* rowPtrG8, int kb) {
  v8bf c0 = *(const v8bf*)(rowPtrG8 + kb);
  v8bf c1 = *(const v8bf*)(rowPtrG8 + kb + 8);
  return __builtin_shufflevector(c0, c1, 0, 1, 2, 3, 4, 5, 6, 7,
                                 8, 9, 10, 11, 12, 13, 14, 15);
}

// ---------------------------------------------------------------- kernel 0: init

__global__ void moe_zero_kernel(float* __restrict__ out, int n, int* __restrict__ cnt) {
  int i = blockIdx.x * blockDim.x + threadIdx.x;
  for (; i < n; i += gridDim.x * blockDim.x) out[i] = 0.0f;
  if (blockIdx.x == 0 && threadIdx.x < NEXP) cnt[threadIdx.x] = 0;
}

// ---------------------------------------------------------------- kernel 0b: x -> bf16

__global__ void moe_cvt_x_kernel(const float* __restrict__ x, __bf16* __restrict__ xbf, int n4) {
  int i = blockIdx.x * blockDim.x + threadIdx.x;
  for (; i < n4; i += gridDim.x * blockDim.x) {
    float4 v = *(const float4*)(x + i * 4);
    __bf16* o = xbf + i * 4;
    o[0] = (__bf16)v.x; o[1] = (__bf16)v.y; o[2] = (__bf16)v.z; o[3] = (__bf16)v.w;
  }
}

// ---------------------------------------------------------------- kernel 1: router
// One wave32 per token; lane e computes logit[t,e]; wave-wide softmax + top-4.

__global__ void moe_router_kernel(const float* __restrict__ x,
                                  const float* __restrict__ rw,
                                  int* __restrict__ cnt,
                                  int* __restrict__ tokIdx,
                                  float* __restrict__ tokW) {
  const int lane = threadIdx.x & 31;
  const int wave = threadIdx.x >> 5;
  const int t = blockIdx.x * (blockDim.x >> 5) + wave;
  if (t >= NTOK) return;

  const float* xr = x + (size_t)t * DH;
  const float* er = rw + (size_t)lane * DH;
  float acc = 0.0f;
  for (int k = 0; k < DH; k += 4) {
    float4 xv = *(const float4*)(xr + k);
    float4 wv = *(const float4*)(er + k);
    acc = fmaf(xv.x, wv.x, acc);
    acc = fmaf(xv.y, wv.y, acc);
    acc = fmaf(xv.z, wv.z, acc);
    acc = fmaf(xv.w, wv.w, acc);
  }

  // softmax over the 32 lanes (fp32)
  float m = acc;
#pragma unroll
  for (int off = 16; off; off >>= 1) m = fmaxf(m, __shfl_xor(m, off, 32));
  float p = __expf(acc - m);
  float s = p;
#pragma unroll
  for (int off = 16; off; off >>= 1) s += __shfl_xor(s, off, 32);
  float prob = p / s;

  // top-4 with lowest-index tie-break, then renormalize
  float v = prob;
  int   selE[NTOPK];
  float selW[NTOPK];
  float wsum = 0.0f;
#pragma unroll
  for (int k = 0; k < NTOPK; ++k) {
    float bv = v;
    int   bi = lane;
#pragma unroll
    for (int off = 16; off; off >>= 1) {
      float ov = __shfl_xor(bv, off, 32);
      int   oi = __shfl_xor(bi, off, 32);
      if (ov > bv || (ov == bv && oi < bi)) { bv = ov; bi = oi; }
    }
    selE[k] = bi;
    selW[k] = bv;
    wsum += bv;
    if (lane == bi) v = -1.0f;  // remove winner
  }

  if (lane == 0) {
    float inv = 1.0f / wsum;
#pragma unroll
    for (int k = 0; k < NTOPK; ++k) {
      int e = selE[k];
      int slot = atomicAdd(&cnt[e], 1);
      tokIdx[e * NTOK + slot] = t;
      tokW[e * NTOK + slot] = selW[k] * inv;
    }
  }
}

// ---------------------------------------------------------------- kernel 2: experts
// grid = (64 token-tiles, 32 experts), block = 256 (8 wave32).
// 2-deep software pipeline: stage-(n+1) raw-B loads issue before stage-n cvt+WMMA,
// with disjoint register sets so loads overlap compute and WMMA WAR gaps close.

__global__ __launch_bounds__(256)
void moe_expert_kernel(const __bf16* __restrict__ xbf,
                       const float* __restrict__ gate_w,
                       const float* __restrict__ up_w,
                       const float* __restrict__ down_w,
                       const int* __restrict__ cnt,
                       const int* __restrict__ tokIdx,
                       const float* __restrict__ tokW,
                       float* __restrict__ out) {
  __shared__ int   tokLds[16];
  __shared__ float wLds[16];
  __shared__ __align__(16) __bf16 hbuf[16 * LDH];  // 16 x 768 (padded) bf16

  const int e = blockIdx.y;
  const int tile = blockIdx.x;
  const int count = cnt[e];
  if (tile * 16 >= count) return;  // uniform early-out (EXEC stays all-ones)

  const int tid = threadIdx.x;
  const int lane = tid & 31;
  const int wave = tid >> 5;
  const int lm = lane & 15;  // row (A/C) or col (B/C) within 16x16 tile
  const int lg = lane >> 4;  // half-wave group

  if (tid < 16) {
    int slot = tile * 16 + tid;
    if (slot < count) {
      tokLds[tid] = tokIdx[e * NTOK + slot];
      wLds[tid] = tokW[e * NTOK + slot];
    } else {
      tokLds[tid] = 0;     // valid dummy row
      wLds[tid] = 0.0f;    // contributes nothing
    }
  }
  __syncthreads();

  // ---------------- phase 1: h = silu(x*gw^T) * (x*uw^T), K = 2048 -------------
  const __bf16* arow = xbf + (size_t)tokLds[lm] * DH + lg * 16;
  const float* gbase = gate_w + (size_t)e * DI * DH;
  const float* ubase = up_w + (size_t)e * DI * DH;

#pragma unroll
  for (int pass = 0; pass < 3; ++pass) {
    const int nt0 = wave * 6 + pass * 2;  // 48 N-tiles over I=768, 2 per pass
    const float* g0 = gbase + (size_t)((nt0 + 0) * 16 + lm) * DH + lg * 16;
    const float* g1 = gbase + (size_t)((nt0 + 1) * 16 + lm) * DH + lg * 16;
    const float* u0 = ubase + (size_t)((nt0 + 0) * 16 + lm) * DH + lg * 16;
    const float* u1 = ubase + (size_t)((nt0 + 1) * 16 + lm) * DH + lg * 16;
    __builtin_prefetch(g0, 0, 3);  // global_prefetch_b8: warm the streams
    __builtin_prefetch(g1, 0, 3);
    __builtin_prefetch(u0, 0, 3);
    __builtin_prefetch(u1, 0, 3);

    v8f aG0 = v8f{}, aG1 = v8f{}, aU0 = v8f{}, aU1 = v8f{};

    // pipeline prologue: stage-0 raw loads
    RawB rg0 = loadB_raw(g0, 0), rg1 = loadB_raw(g1, 0);
    RawB ru0 = loadB_raw(u0, 0), ru1 = loadB_raw(u1, 0);
    v16bf af = fragA_bf16(arow, 0);

    for (int kb = 0; kb < DH; kb += 64) {
      // issue odd-stage loads (kb+32) into the second register set
      RawB sg0 = loadB_raw(g0, kb + 32), sg1 = loadB_raw(g1, kb + 32);
      RawB su0 = loadB_raw(u0, kb + 32), su1 = loadB_raw(u1, kb + 32);
      v16bf af2 = fragA_bf16(arow, kb + 32);

      // compute even stage (loads were issued one stage earlier)
      aG0 = wmma_bf16(af, cvtB(rg0), aG0);
      aG1 = wmma_bf16(af, cvtB(rg1), aG1);
      aU0 = wmma_bf16(af, cvtB(ru0), aU0);
      aU1 = wmma_bf16(af, cvtB(ru1), aU1);

      // issue next even-stage loads (kb+64)
      int kn = kb + 64;
      if (kn < DH) {
        rg0 = loadB_raw(g0, kn); rg1 = loadB_raw(g1, kn);
        ru0 = loadB_raw(u0, kn); ru1 = loadB_raw(u1, kn);
        af = fragA_bf16(arow, kn);
      }

      // compute odd stage
      aG0 = wmma_bf16(af2, cvtB(sg0), aG0);
      aG1 = wmma_bf16(af2, cvtB(sg1), aG1);
      aU0 = wmma_bf16(af2, cvtB(su0), aU0);
      aU1 = wmma_bf16(af2, cvtB(su1), aU1);
    }

    // epilogue: silu(gate) * up -> bf16 LDS tile (C/D layout: m = lg*8+r, n = lm)
#pragma unroll
    for (int r = 0; r < 8; ++r) {
      int mrow = lg * 8 + r;
      float gA = aG0[r], uA = aU0[r];
      float gB = aG1[r], uB = aU1[r];
      float hA = (gA / (1.0f + __expf(-gA))) * uA;
      float hB = (gB / (1.0f + __expf(-gB))) * uB;
      hbuf[mrow * LDH + (nt0 + 0) * 16 + lm] = (__bf16)hA;
      hbuf[mrow * LDH + (nt0 + 1) * 16 + lm] = (__bf16)hB;
    }
  }
  __syncthreads();

  // ---------------- phase 2: y = h * dw^T, K = 768, scatter-add ----------------
  const __bf16* hrow = hbuf + lm * LDH + lg * 16;
  const float* dbase = down_w + (size_t)e * DH * DI;

#pragma unroll
  for (int pass = 0; pass < 8; ++pass) {
    const int nt0 = wave * 16 + pass * 2;  // 128 N-tiles over H=2048, 2 per pass
    const float* d0 = dbase + (size_t)((nt0 + 0) * 16 + lm) * DI + lg * 16;
    const float* d1 = dbase + (size_t)((nt0 + 1) * 16 + lm) * DI + lg * 16;
    __builtin_prefetch(d0, 0, 3);
    __builtin_prefetch(d1, 0, 3);

    v8f aD0 = v8f{}, aD1 = v8f{};

    RawB rd0 = loadB_raw(d0, 0), rd1 = loadB_raw(d1, 0);

    for (int kb = 0; kb < DI; kb += 64) {
      RawB sd0 = loadB_raw(d0, kb + 32), sd1 = loadB_raw(d1, kb + 32);

      v16bf a0 = fragA_bf16(hrow, kb);  // LDS: low latency, no pipelining needed
      aD0 = wmma_bf16(a0, cvtB(rd0), aD0);
      aD1 = wmma_bf16(a0, cvtB(rd1), aD1);

      int kn = kb + 64;
      if (kn < DI) { rd0 = loadB_raw(d0, kn); rd1 = loadB_raw(d1, kn); }

      v16bf a1 = fragA_bf16(hrow, kb + 32);
      aD0 = wmma_bf16(a1, cvtB(sd0), aD0);
      aD1 = wmma_bf16(a1, cvtB(sd1), aD1);
    }

    // scale by router weight and atomically accumulate into out
#pragma unroll
    for (int r = 0; r < 8; ++r) {
      int mrow = lg * 8 + r;
      float wv = wLds[mrow];
      size_t orow = (size_t)tokLds[mrow] * DH;
      atomicAdd(&out[orow + (nt0 + 0) * 16 + lm], wv * aD0[r]);
      atomicAdd(&out[orow + (nt0 + 1) * 16 + lm], wv * aD1[r]);
    }
  }
}

// ---------------------------------------------------------------- launcher

extern "C" void kernel_launch(void* const* d_in, const int* in_sizes, int n_in,
                              void* d_out, int out_size, void* d_ws, size_t ws_size,
                              hipStream_t stream) {
  const float* x       = (const float*)d_in[0];  // [1,1024,2048]
  const float* routerW = (const float*)d_in[1];  // [32,2048]
  const float* gateW   = (const float*)d_in[2];  // [32,768,2048]
  const float* upW     = (const float*)d_in[3];  // [32,768,2048]
  const float* downW   = (const float*)d_in[4];  // [32,2048,768]
  float* out = (float*)d_out;                    // [1,1024,2048]

  // workspace layout: bf16 activations first (16B-aligned base), then routing
  __bf16* xbf   = (__bf16*)d_ws;                    // [1024][2048] bf16 = 4 MB
  int*    cnt   = (int*)(xbf + (size_t)NTOK * DH);  // [32]
  int*    tokIdx = cnt + NEXP;                      // [32][1024]
  float*  tokW  = (float*)(tokIdx + NEXP * NTOK);   // [32][1024]

  // 1) zero output accumulator + expert counters
  moe_zero_kernel<<<1024, 256, 0, stream>>>(out, NTOK * DH, cnt);

  // 1b) convert activations to bf16 (A operand is re-read many times; B streams fp32)
  moe_cvt_x_kernel<<<512, 256, 0, stream>>>(x, xbf, NTOK * DH / 4);

  // 2) router: softmax + top-4 + build per-expert token lists
  moe_router_kernel<<<NTOK / 8, 256, 0, stream>>>(x, routerW, cnt, tokIdx, tokW);

  // 3) expert FFN: one workgroup per (16-token tile, expert)
  dim3 grid(NTOK / 16, NEXP);
  moe_expert_kernel<<<grid, 256, 0, stream>>>(xbf, gateW, upW, downW,
                                              cnt, tokIdx, tokW, out);
}
